// GNN_54906861912269
// MI455X (gfx1250) — matrix operand order repaired and checked by reading, not compiled, not run on hardware
//
#include <hip/hip_runtime.h>

#define NN   50000
#define EE   800000
#define CC   128
#define HH   2
#define HCC  256          // H*C
#define ET   (EE + NN)    // edges + self loops

// ---------------- bf16 helpers / WMMA types ----------------
typedef __attribute__((ext_vector_type(16))) __bf16 v16bf;
typedef __attribute__((ext_vector_type(8)))  float  v8f;

__device__ __forceinline__ unsigned short f2bf(float f) {
    unsigned int u = __float_as_uint(f);
    return (unsigned short)((u + 0x7FFFu + ((u >> 16) & 1u)) >> 16);
}

// order-preserving float <-> uint encoding for atomicMax-based segment max
__device__ __forceinline__ unsigned int fenc(float f) {
    unsigned int u = __float_as_uint(f);
    return (u & 0x80000000u) ? ~u : (u | 0x80000000u);
}
__device__ __forceinline__ float fdec(unsigned int u) {
    u = (u & 0x80000000u) ? (u & 0x7FFFFFFFu) : ~u;
    return __uint_as_float(u);
}
#define ENC_NEG_INF 0x007FFFFFu   // fenc(-inf)

__device__ __forceinline__ float lrelu(float v) { return v > 0.0f ? v : 0.2f * v; }

// ---------------- degree + self-loop attr ----------------
__global__ void deg_kernel(const int* __restrict__ dst, const float* __restrict__ eattr,
                           float* deg, float* loopa, int E) {
    int e = blockIdx.x * blockDim.x + threadIdx.x;
    if (e >= E) return;
    int d = dst[e];
    atomicAdd(&deg[d], 1.0f);
    atomicAdd(&loopa[d], eattr[e]);
}

__global__ void loopdiv_kernel(float* loopa, const float* __restrict__ deg, int n) {
    int i = blockIdx.x * blockDim.x + threadIdx.x;
    if (i >= n) return;
    loopa[i] = loopa[i] / fmaxf(deg[i], 1.0f);
}

// ---------------- fp32 -> bf16 conversion ----------------
__global__ void cvt_x_kernel(const float* __restrict__ x, unsigned short* xbf, int total) {
    int i = blockIdx.x * blockDim.x + threadIdx.x;
    if (i >= total) return;
    xbf[i] = f2bf(x[i]);
}

// Pre-pack W (128x256 fp32, row-major) directly into WMMA B-fragment order:
//   idx = ((which*16 + ntile)*4 + kstep)*32 + lane)*16 + e
//   where e -> (j = e>>1, parity = e&1), k = kstep*32 + 2j + 16*(lane>>4) + parity,
//         n = ntile*16 + (lane&15)
// so each lane's 16 bf16 B-fragment elements are CONTIGUOUS (two b128 loads).
__global__ void cvt_w_kernel(const float* __restrict__ Wl, const float* __restrict__ Wr,
                             unsigned short* __restrict__ wpack) {
    int i = blockIdx.x * blockDim.x + threadIdx.x;
    if (i >= 2 * CC * HCC) return;
    int e     = i & 15;
    int lane  = (i >> 4) & 31;
    int ks    = (i >> 9) & 3;
    int ntile = (i >> 11) & 15;
    int which = i >> 15;
    int k = ks * 32 + ((e >> 1) << 1) + ((lane >> 4) << 4) + (e & 1);
    int n = ntile * 16 + (lane & 15);
    const float* W = which ? Wr : Wl;
    wpack[i] = f2bf(W[k * HCC + n]);
}

// ---------------- WMMA GEMM: Y[N,256] = Xbf[N,128] @ W[128,256] + bias ----------------
// One wave owns one 16-wide N-tile and 4 consecutive 16-row M-tiles.
// B fragments for all 4 K-steps held in registers (reused across the 4 M-tiles);
// A fragments arrive as two b128 loads per K-step (this code's contiguous k = 16*half + s order, same as packed B:
// dword offsets form two contiguous 4-dword runs). 16 v_wmma per wave iteration.
__global__ void __launch_bounds__(256) gemm_kernel(const unsigned short* __restrict__ Xbf,
                                                   const uint4* __restrict__ Wpack,
                                                   const float* __restrict__ bias,
                                                   float* __restrict__ Y) {
    int wave  = threadIdx.x >> 5;
    int lane  = threadIdx.x & 31;
    int ntile = blockIdx.y * 8 + wave;
    int half  = lane >> 4;
    int col   = lane & 15;

    union U { v16bf v; unsigned int d[8]; uint4 q[2]; };

    // Load all 4 K-step B fragments once (contiguous, prepacked).
    U b[4];
#pragma unroll
    for (int ks = 0; ks < 4; ++ks) {
        const uint4* bp = Wpack + ((size_t)(ntile * 4 + ks) * 32 + lane) * 2;
        b[ks].q[0] = bp[0];
        b[ks].q[1] = bp[1];
    }
    float bval = bias[ntile * 16 + col];

#pragma unroll
    for (int t = 0; t < 4; ++t) {
        int m0 = (blockIdx.x * 4 + t) * 16;
        if (m0 >= NN) break;                       // wave-uniform: EXEC stays all-ones
        const uint4* Xrow = (const uint4*)(Xbf + (size_t)(m0 + col) * CC);
        v8f c = {};
#pragma unroll
        for (int ks = 0; ks < 4; ++ks) {
            U a;
            a.q[0] = Xrow[4 * ks + 2 * half];      // dwords {0..3}+8*half  of K-step
            a.q[1] = Xrow[4 * ks + 2 * half + 1];  // dwords {4..7}+8*half of K-step
            c = __builtin_amdgcn_wmma_f32_16x16x32_bf16(false, a.v, false, b[ks].v,
                                                        (short)0, c, false, false);
        }
#pragma unroll
        for (int r = 0; r < 8; ++r) {
            int m = m0 + r + (half << 3);
            Y[(size_t)m * HCC + ntile * 16 + col] = c[r] + bval;
        }
    }
}

// ---------------- per-layer segment-max / denom init ----------------
__global__ void init_kernel(unsigned int* smax, float* denom, int n2) {
    int i = blockIdx.x * blockDim.x + threadIdx.x;
    if (i >= n2) return;
    smax[i] = ENC_NEG_INF;
    denom[i] = 0.0f;
}

// ---------------- attention scores: one wave per edge ----------------
__global__ void score_kernel(const int* __restrict__ src, const int* __restrict__ dst,
                             const float* __restrict__ eattr, const float* __restrict__ loopa,
                             const float* __restrict__ xl, const float* __restrict__ xr,
                             const float* __restrict__ We, const float* __restrict__ att,
                             float* __restrict__ score, unsigned int* smax) {
    int wave = threadIdx.x >> 5;
    int lane = threadIdx.x & 31;
    int e = blockIdx.x * 8 + wave;
    if (e >= ET) return;
    int s, d; float ea;
    if (e < EE) { s = src[e]; d = dst[e]; ea = eattr[e]; }
    else        { s = d = e - EE; ea = loopa[s]; }

    const float* xls = xl + (size_t)s * HCC;
    const float* xrd = xr + (size_t)d * HCC;

#pragma unroll
    for (int h = 0; h < HH; ++h) {
        float p = 0.0f;
#pragma unroll
        for (int t = 0; t < 4; ++t) {
            int cidx = h * CC + lane + 32 * t;
            float m = xls[cidx] + xrd[cidx] + ea * We[cidx];
            p += lrelu(m) * att[cidx];
        }
        for (int o = 16; o > 0; o >>= 1) p += __shfl_xor(p, o);
        if (lane == 0) {
            score[(size_t)e * HH + h] = p;
            atomicMax(&smax[d * HH + h], fenc(p));
        }
    }
}

// ---------------- alpha = exp(score - smax); denom segment sum ----------------
__global__ void alpha_kernel(const int* __restrict__ dst, float* __restrict__ score,
                             const unsigned int* __restrict__ smax, float* denom) {
    int i = blockIdx.x * blockDim.x + threadIdx.x;
    if (i >= ET * HH) return;
    int e = i >> 1, h = i & 1;
    int d = (e < EE) ? dst[e] : (e - EE);
    float a = expf(score[i] - fdec(smax[d * HH + h]));
    score[i] = a;
    atomicAdd(&denom[d * HH + h], a);
}

// ---------------- aggregate: g[d, layer, :] += mean_h(alpha_h/denom * xl[s, h, :]) ----------------
__global__ void agg_kernel(const int* __restrict__ src, const int* __restrict__ dst,
                           const float* __restrict__ score, const float* __restrict__ denom,
                           const float* __restrict__ xl, float* __restrict__ g, int layer) {
    int wave = threadIdx.x >> 5;
    int lane = threadIdx.x & 31;
    int e = blockIdx.x * 8 + wave;
    if (e >= ET) return;
    int s, d;
    if (e < EE) { s = src[e]; d = dst[e]; }
    else        { s = d = e - EE; }

    const float* xls = xl + (size_t)s * HCC;
    float* gd = g + (size_t)d * HCC + layer * CC;

#pragma unroll
    for (int h = 0; h < HH; ++h) {
        float w = 0.5f * score[(size_t)e * HH + h] / denom[d * HH + h];
#pragma unroll
        for (int t = 0; t < 4; ++t) {
            int c = lane + 32 * t;
            atomicAdd(&gd[c], w * xls[h * CC + c]);
        }
    }
}

// ---------------- finalize layer: +bo, leaky_relu ----------------
__global__ void fin_kernel(float* g, const float* __restrict__ bo, int layer) {
    int i = blockIdx.x * blockDim.x + threadIdx.x;
    if (i >= NN * CC) return;
    int n = i >> 7, c = i & 127;
    size_t idx = (size_t)n * HCC + layer * CC + c;
    g[idx] = lrelu(g[idx] + bo[c]);
}

// ---------------- conv1d(K=7, pad 3) over spatial C, 2 in/out channels + softmax over channels ----------------
__global__ void conv_kernel(const float* __restrict__ g, const float* __restrict__ cw,
                            const float* __restrict__ cb, float* __restrict__ out) {
    int i = blockIdx.x * blockDim.x + threadIdx.x;
    if (i >= NN * CC) return;
    int n = i >> 7, c = i & 127;
    const float* gn = g + (size_t)n * HCC;
    float a0 = cb[0], a1 = cb[1];
#pragma unroll
    for (int ii = 0; ii < 2; ++ii) {
#pragma unroll
        for (int k = 0; k < 7; ++k) {
            int cc = c + k - 3;
            if (cc >= 0 && cc < CC) {
                float v = gn[ii * CC + cc];
                a0 += v * cw[ii * 7 + k];
                a1 += v * cw[14 + ii * 7 + k];
            }
        }
    }
    float m = fmaxf(a0, a1);
    float e0 = expf(a0 - m), e1 = expf(a1 - m);
    float inv = 1.0f / (e0 + e1);
    out[(size_t)n * HCC + c]      = e0 * inv;
    out[(size_t)n * HCC + CC + c] = e1 * inv;
}

// ---------------- host launcher ----------------
extern "C" void kernel_launch(void* const* d_in, const int* in_sizes, int n_in,
                              void* d_out, int out_size, void* d_ws, size_t ws_size,
                              hipStream_t stream) {
    const float* x      = (const float*)d_in[0];
    const int*   ei     = (const int*)d_in[1];
    const float* eattr  = (const float*)d_in[2];
    const float* conv_w = (const float*)d_in[17];
    const float* conv_b = (const float*)d_in[18];
    const int* srcp = ei;
    const int* dstp = ei + EE;

    // workspace carve-out (256B aligned)
    char* ws = (char*)d_ws;
    size_t off = 0;
    auto take = [&](size_t bytes) -> char* {
        char* p = ws + off;
        off += (bytes + 255) & ~(size_t)255;
        return p;
    };
    unsigned short* xbf  = (unsigned short*)take((size_t)NN * CC * 2);
    unsigned short* wbf  = (unsigned short*)take((size_t)2 * CC * HCC * 2);
    float* xl    = (float*)take((size_t)NN * HCC * 4);
    float* xr    = (float*)take((size_t)NN * HCC * 4);
    float* deg   = (float*)take((size_t)NN * 4);
    float* loopa = (float*)take((size_t)NN * 4);
    float* score = (float*)take((size_t)ET * HH * 4);
    unsigned int* smax = (unsigned int*)take((size_t)NN * HH * 4);
    float* denom = (float*)take((size_t)NN * HH * 4);
    float* g     = (float*)take((size_t)NN * HCC * 4);

    hipMemsetAsync(deg,   0, (size_t)NN * 4, stream);
    hipMemsetAsync(loopa, 0, (size_t)NN * 4, stream);
    hipMemsetAsync(g,     0, (size_t)NN * HCC * 4, stream);

    deg_kernel<<<(EE + 255) / 256, 256, 0, stream>>>(dstp, eattr, deg, loopa, EE);
    loopdiv_kernel<<<(NN + 255) / 256, 256, 0, stream>>>(loopa, deg, NN);
    cvt_x_kernel<<<(NN * CC + 255) / 256, 256, 0, stream>>>(x, xbf, NN * CC);

    const int MT = (NN / 16 + 3) / 4;   // 3125 M-tiles, 4 per wave-iteration

    for (int L = 0; L < 2; ++L) {
        const float* Wl  = (const float*)d_in[3 + 7 * L];
        const float* bl  = (const float*)d_in[4 + 7 * L];
        const float* Wr  = (const float*)d_in[5 + 7 * L];
        const float* br  = (const float*)d_in[6 + 7 * L];
        const float* We  = (const float*)d_in[7 + 7 * L];
        const float* att = (const float*)d_in[8 + 7 * L];
        const float* bo  = (const float*)d_in[9 + 7 * L];

        cvt_w_kernel<<<(2 * CC * HCC + 255) / 256, 256, 0, stream>>>(Wl, Wr, wbf);

        dim3 ggrid(MT, 2);
        gemm_kernel<<<ggrid, 256, 0, stream>>>(xbf, (const uint4*)wbf, bl, xl);
        gemm_kernel<<<ggrid, 256, 0, stream>>>(xbf, (const uint4*)(wbf + (size_t)CC * HCC),
                                               br, xr);

        init_kernel<<<(NN * HH + 255) / 256, 256, 0, stream>>>(smax, denom, NN * HH);
        score_kernel<<<(ET + 7) / 8, 256, 0, stream>>>(srcp, dstp, eattr, loopa,
                                                       xl, xr, We, att, score, smax);
        alpha_kernel<<<(ET * HH + 255) / 256, 256, 0, stream>>>(dstp, score, smax, denom);
        agg_kernel<<<(ET + 7) / 8, 256, 0, stream>>>(srcp, dstp, score, denom, xl, g, L);
        fin_kernel<<<(NN * CC + 255) / 256, 256, 0, stream>>>(g, bo, L);
    }

    conv_kernel<<<(NN * CC + 255) / 256, 256, 0, stream>>>(g, conv_w, conv_b, (float*)d_out);
}
